// PartitionAttentionCl_9972914061552
// MI455X (gfx1250) — matrix operand order repaired and checked by reading, not compiled, not run on hardware
//
#include <hip/hip_runtime.h>

// ---- problem constants -------------------------------------------------
#define C_     256
#define DH_    32
#define NH_    8
#define NTOK_  49      // 7*7 tokens per window
#define NP_    64      // padded window rows
#define HW_    56
#define WINS_  2048    // 32 * 8 * 8
#define NROWS_ 100352  // 32*56*56

typedef __attribute__((ext_vector_type(16))) __bf16 v16bf;
typedef __attribute__((ext_vector_type(8)))  float  v8f;

__device__ __forceinline__ unsigned short f2bf(float f) {
    union { float f; unsigned u; } x; x.f = f;
    unsigned u = x.u + 0x7FFFu + ((x.u >> 16) & 1u);   // round-to-nearest-even
    return (unsigned short)(u >> 16);
}

__device__ __forceinline__ float wredsum(float v) {
#pragma unroll
    for (int o = 16; o > 0; o >>= 1) v += __shfl_xor(v, o, 32);
    return v;
}

// scalar wave id (keeps per-wave control flow on the SALU)
__device__ __forceinline__ int wave_id() {
    return __builtin_amdgcn_readfirstlane((int)(threadIdx.x >> 5));
}

// async global->LDS staging (GLOBAL_LOAD_ASYNC_TO_LDS_B128, ASYNCcnt) with
// a VGPR round-trip fallback if the builtin is unavailable.
__device__ __forceinline__ void lds_stage(unsigned short* dst, const unsigned short* src,
                                          int nchunks, int tid, int nthr) {
#if __has_builtin(__builtin_amdgcn_global_load_async_to_lds_b128)
    typedef __attribute__((__vector_size__(4 * sizeof(int)))) int vsi4;
    __attribute__((address_space(1))) vsi4* gp =
        (__attribute__((address_space(1))) vsi4*)(size_t)src;
    __attribute__((address_space(3))) vsi4* lp =
        (__attribute__((address_space(3))) vsi4*)dst;
    for (int i = tid; i < nchunks; i += nthr)
        __builtin_amdgcn_global_load_async_to_lds_b128(gp + i, lp + i, 0, 0);
#if __has_builtin(__builtin_amdgcn_s_wait_asynccnt)
    __builtin_amdgcn_s_wait_asynccnt(0);
#else
    asm volatile("s_wait_asynccnt 0x0" ::: "memory");
#endif
#else
    const uint4* s4 = (const uint4*)src;
    uint4* d4 = (uint4*)dst;
    for (int i = tid; i < nchunks; i += nthr) d4[i] = s4[i];
#endif
}

// A-fragment (16x32 bf16) from row-major ushort buffer, leading dim ld.
// Layout: lanes 0-15: M=lane, K=0..15; lanes 16-31: M=lane-16,
// K=16..31; each VGPR holds a consecutive (K,K+1) bf16 pair.
__device__ __forceinline__ v16bf fragA(const unsigned short* base, int ld,
                                       int mBase, int kBase) {
    int lane = threadIdx.x & 31;
    int half = lane >> 4;
    const unsigned short* p = base + (size_t)(mBase + (lane & 15)) * ld + kBase + half * 16;
    union { v16bf v; unsigned u[8]; } r;
#pragma unroll
    for (int i = 0; i < 4; i++) r.u[i]     = *(const unsigned*)(p + 2 * i);
#pragma unroll
    for (int i = 0; i < 4; i++) r.u[4 + i] = *(const unsigned*)(p + 8 + 2 * i);
    return r.v;
}

// B-fragment (32x16 bf16) from a K-major (transposed) buffer: element (k,n) at
// base[(nBase+n)*ldk + k]. Contiguous-k layout (same as fragA): lanes 0-15: N=lane, K=0..15 (2/VGPR);
// lanes 16-31: N=lane-16, K=16..31.
__device__ __forceinline__ v16bf fragBt(const unsigned short* base, int ldk,
                                        int nBase, int kBase) {
    int lane = threadIdx.x & 31;
    int half = lane >> 4;
    const unsigned short* p = base + (size_t)(nBase + (lane & 15)) * ldk + kBase + half * 16;
    union { v16bf v; unsigned u[8]; } r;
#pragma unroll
    for (int i = 0; i < 8; i++) r.u[i] = *(const unsigned*)(p + 2 * i);
    return r.v;
}

#define WMMA_BF16(acc, a, b) \
    acc = __builtin_amdgcn_wmma_f32_16x16x32_bf16(false, a, false, b, (short)0, acc, false, false)

// ---- K0: convert + transpose weights to bf16 (K-major) -----------------
__global__ void k_convert_weights(const float* __restrict__ qkv_w,
                                  const float* __restrict__ proj_w,
                                  const float* __restrict__ fc1_w,
                                  const float* __restrict__ fc2_w,
                                  unsigned short* qkvT, unsigned short* projT,
                                  unsigned short* fc1T, unsigned short* fc2T) {
    int stride = gridDim.x * blockDim.x;
    int tid = blockIdx.x * blockDim.x + threadIdx.x;
    for (int i = tid; i < 768 * 256; i += stride) {            // qkv_w [256][768] -> [768][256]
        int n = i / 256, k = i % 256;
        qkvT[i] = f2bf(qkv_w[k * 768 + n]);
    }
    for (int i = tid; i < 256 * 256; i += stride) {            // proj_w [256][256] -> transposed
        int n = i / 256, k = i % 256;
        projT[i] = f2bf(proj_w[k * 256 + n]);
    }
    for (int i = tid; i < 1024 * 256; i += stride) {           // fc1_w [256][1024] -> [1024][256]
        int n = i / 256, k = i % 256;
        fc1T[i] = f2bf(fc1_w[k * 1024 + n]);
    }
    for (int i = tid; i < 256 * 1024; i += stride) {           // fc2_w [1024][256] -> [256][1024]
        int n = i / 1024, k = i % 1024;
        fc2T[i] = f2bf(fc2_w[k * 256 + n]);
    }
}

// ---- K1: LN1 + window partition, fp32 -> bf16, pad rows 49..63 ---------
__global__ void __launch_bounds__(256) k_ln1_partition(const float* __restrict__ x,
                                                       const float* __restrict__ w,
                                                       const float* __restrict__ b,
                                                       unsigned short* __restrict__ xw) {
    int win = blockIdx.x;
    int bi = win >> 6, wr = (win >> 3) & 7, wc = win & 7;
    int wave = wave_id(), lane = threadIdx.x & 31;
    unsigned short* dst = xw + (size_t)win * NP_ * C_;
    for (int r = wave; r < NTOK_; r += 8) {
        int i = r / 7, j = r % 7;
        const float* row = x + (((size_t)bi * HW_ + wr * 7 + i) * HW_ + wc * 7 + j) * C_;
        float4 v0 = *(const float4*)(row + lane * 8);
        float4 v1 = *(const float4*)(row + lane * 8 + 4);
        float vv[8] = {v0.x, v0.y, v0.z, v0.w, v1.x, v1.y, v1.z, v1.w};
        float s = 0.f, sq = 0.f;
#pragma unroll
        for (int t = 0; t < 8; t++) { s += vv[t]; sq += vv[t] * vv[t]; }
        s = wredsum(s); sq = wredsum(sq);
        float mu  = s * (1.0f / 256.0f);
        float var = sq * (1.0f / 256.0f) - mu * mu;
        float rs  = rsqrtf(var + 1e-6f);
        int c0 = lane * 8;
        unsigned short* o = dst + r * C_ + c0;
#pragma unroll
        for (int t = 0; t < 8; t++)
            o[t] = f2bf((vv[t] - mu) * rs * w[c0 + t] + b[c0 + t]);
    }
    for (int idx = threadIdx.x; idx < (NP_ - NTOK_) * C_; idx += blockDim.x)
        dst[NTOK_ * C_ + idx] = 0;
}

// ---- K2: fused per-(window,head) QKV GEMM + attention ------------------
__global__ void __launch_bounds__(128) k_window_attn(const unsigned short* __restrict__ xw,
                                                     const unsigned short* __restrict__ qkvT,
                                                     const float* __restrict__ qkv_b,
                                                     const float* __restrict__ bias_table,
                                                     unsigned short* __restrict__ ow) {
    __shared__ __align__(16) unsigned char  smemA[NP_ * C_ * 2];   // 32KB: As, later Ss+Ps
    __shared__ __align__(16) unsigned short Qs[NP_ * DH_];         // 4KB (scaled Q)
    __shared__ __align__(16) unsigned short Ks[NP_ * DH_];         // 4KB
    __shared__ __align__(16) unsigned short Vt[DH_ * NP_];         // 4KB (V transposed)
    int win = blockIdx.x, head = blockIdx.y;
    int wave = wave_id(), lane = threadIdx.x & 31;
    int half = lane >> 4, nn = lane & 15;
    unsigned short* As = (unsigned short*)smemA;

    __builtin_prefetch(qkvT + (size_t)head * 96 * C_, 0, 1);

    // stage LN1'd window tile (64x256 bf16) into LDS via async-to-LDS
    lds_stage(As, xw + (size_t)win * NP_ * C_, NP_ * C_ / 8, threadIdx.x, 128);
    __syncthreads();

    const float scale = 0.17677669529663687f;   // 1/sqrt(32)
    // QKV GEMM: wave owns M-stripe `wave`; A-fragments hoisted (loop-invariant)
    v16bf afr[8];
#pragma unroll
    for (int kb = 0; kb < 8; kb++) afr[kb] = fragA(As, C_, wave * 16, kb * 32);
    for (int nt = 0; nt < 6; nt++) {
        int colBase = head * 96 + nt * 16;
        v8f acc = {};
#pragma unroll
        for (int kb = 0; kb < 8; kb++) {
            v16bf bm = fragBt(qkvT, C_, colBase, kb * 32);
            WMMA_BF16(acc, afr[kb], bm);
        }
#pragma unroll
        for (int r = 0; r < 8; r++) {
            int row = wave * 16 + r + 8 * half;
            float v = acc[r] + qkv_b[colBase + nn];
            if (nt < 2)      Qs[row * DH_ + nt * 16 + nn] = f2bf(v * scale);
            else if (nt < 4) Ks[row * DH_ + (nt - 2) * 16 + nn] = f2bf(v);
            else             Vt[((nt - 4) * 16 + nn) * NP_ + row] = f2bf(v);
        }
    }
    __syncthreads();

    float*          Ss = (float*)smemA;                            // 16KB
    unsigned short* Ps = (unsigned short*)(smemA + NP_ * NP_ * 4); // 8KB

    // S = (Q*scale) @ K^T + rel_bias ; single K=32 WMMA per 16x16 tile
    v16bf qfr = fragA(Qs, DH_, wave * 16, 0);
    for (int nt = 0; nt < 4; nt++) {
        v16bf bm = fragBt(Ks, DH_, nt * 16, 0);
        v8f acc = {};
        WMMA_BF16(acc, qfr, bm);
#pragma unroll
        for (int r = 0; r < 8; r++) {
            int nq = wave * 16 + r + 8 * half;
            int mk = nt * 16 + nn;
            float v;
            if (mk >= NTOK_)      v = -1e30f;
            else if (nq >= NTOK_) v = acc[r];
            else {
                int di = nq / 7 - mk / 7 + 6;
                int dj = nq % 7 - mk % 7 + 6;
                v = acc[r] + bias_table[(di * 13 + dj) * NH_ + head];
            }
            Ss[nq * NP_ + mk] = v;
        }
    }
    __syncthreads();

    // softmax: one thread per row (two-pass over LDS)
    if (threadIdx.x < NP_) {
        float* row = Ss + threadIdx.x * NP_;
        float mx = -1e30f;
        for (int m = 0; m < NTOK_; m++) mx = fmaxf(mx, row[m]);
        float sum = 0.f;
        for (int m = 0; m < NTOK_; m++) sum += __expf(row[m] - mx);
        float inv = 1.0f / sum;
        unsigned short* prow = Ps + threadIdx.x * NP_;
        for (int m = 0; m < NTOK_; m++) prow[m] = f2bf(__expf(row[m] - mx) * inv);
        for (int m = NTOK_; m < NP_; m++) prow[m] = 0;
    }
    __syncthreads();

    // O = P @ V  (64x32): wave owns M-stripe; K=64 -> 2 WMMA steps
    unsigned short* owp = ow + (size_t)win * NP_ * C_ + head * DH_;
    v16bf pfr0 = fragA(Ps, NP_, wave * 16, 0);
    v16bf pfr1 = fragA(Ps, NP_, wave * 16, 32);
    for (int nt = 0; nt < 2; nt++) {
        v8f acc = {};
        WMMA_BF16(acc, pfr0, fragBt(Vt, NP_, nt * 16, 0));
        WMMA_BF16(acc, pfr1, fragBt(Vt, NP_, nt * 16, 32));
#pragma unroll
        for (int r = 0; r < 8; r++) {
            int row = wave * 16 + r + 8 * half;
            owp[row * C_ + nt * 16 + nn] = f2bf(acc[r]);
        }
    }
}

// ---- K3: proj GEMM + window-reverse + residual -> d_out ----------------
__global__ void __launch_bounds__(128) k_proj(const unsigned short* __restrict__ ow,
                                              const unsigned short* __restrict__ projT,
                                              const float* __restrict__ proj_b,
                                              const float* __restrict__ x,
                                              float* __restrict__ out) {
    __shared__ __align__(16) unsigned short Os[NP_ * C_];   // 32KB
    int win = blockIdx.x;
    int bi = win >> 6, wr = (win >> 3) & 7, wc = win & 7;
    int wave = wave_id(), lane = threadIdx.x & 31;
    int half = lane >> 4, nn = lane & 15;

    __builtin_prefetch(projT, 0, 1);
    lds_stage(Os, ow + (size_t)win * NP_ * C_, NP_ * C_ / 8, threadIdx.x, 128);
    __syncthreads();

    v16bf afr[8];
#pragma unroll
    for (int kb = 0; kb < 8; kb++) afr[kb] = fragA(Os, C_, wave * 16, kb * 32);

    for (int nt = 0; nt < 16; nt++) {
        v8f acc = {};
#pragma unroll
        for (int kb = 0; kb < 8; kb++) {
            v16bf bm = fragBt(projT, C_, nt * 16, kb * 32);
            WMMA_BF16(acc, afr[kb], bm);
        }
#pragma unroll
        for (int r = 0; r < 8; r++) {
            int tok = wave * 16 + r + 8 * half;
            if (tok < NTOK_) {
                int i = tok / 7, j = tok % 7;
                int col = nt * 16 + nn;
                size_t gi = (((size_t)bi * HW_ + wr * 7 + i) * HW_ + wc * 7 + j) * C_ + col;
                out[gi] = x[gi] + acc[r] + proj_b[col];
            }
        }
    }
}

// ---- K4: LN2 + MLP (fc1/GELU/fc2) + residual, in place on d_out --------
__global__ void __launch_bounds__(128) k_mlp(float* __restrict__ out,
                                             const float* __restrict__ n2w,
                                             const float* __restrict__ n2b,
                                             const unsigned short* __restrict__ fc1T,
                                             const float* __restrict__ fc1_b,
                                             const unsigned short* __restrict__ fc2T,
                                             const float* __restrict__ fc2_b) {
    __shared__ __align__(16) unsigned short As[64 * C_];    // 32KB: LN2 activations
    __shared__ __align__(16) unsigned short Hs[64 * 128];   // 16KB: fc1 chunk
    size_t rowBase = (size_t)blockIdx.x * 64;
    int wave = wave_id(), lane = threadIdx.x & 31;
    int half = lane >> 4, nn = lane & 15;

    for (int r = wave; r < 64; r += 4) {
        const float* row = out + (rowBase + r) * C_;
        float4 v0 = *(const float4*)(row + lane * 8);
        float4 v1 = *(const float4*)(row + lane * 8 + 4);
        float vv[8] = {v0.x, v0.y, v0.z, v0.w, v1.x, v1.y, v1.z, v1.w};
        float s = 0.f, sq = 0.f;
#pragma unroll
        for (int t = 0; t < 8; t++) { s += vv[t]; sq += vv[t] * vv[t]; }
        s = wredsum(s); sq = wredsum(sq);
        float mu  = s * (1.0f / 256.0f);
        float var = sq * (1.0f / 256.0f) - mu * mu;
        float rs  = rsqrtf(var + 1e-6f);
        int c0 = lane * 8;
#pragma unroll
        for (int t = 0; t < 8; t++)
            As[r * C_ + c0 + t] = f2bf((vv[t] - mu) * rs * n2w[c0 + t] + n2b[c0 + t]);
    }
    __syncthreads();

    v8f zero = {};
    v8f acc2[16];
#pragma unroll
    for (int i = 0; i < 16; i++) acc2[i] = zero;

    for (int c = 0; c < 8; c++) {       // 128-wide fc1 column chunks
        // gemm1: H_chunk = gelu(A @ fc1T[:,chunk] + b1); wave owns M-stripe `wave`
        v16bf afr1[8];
#pragma unroll
        for (int kb = 0; kb < 8; kb++) afr1[kb] = fragA(As, C_, wave * 16, kb * 32);
        for (int nt = 0; nt < 8; nt++) {
            v8f acc = {};
#pragma unroll
            for (int kb = 0; kb < 8; kb++) {
                v16bf bm = fragBt(fc1T, C_, c * 128 + nt * 16, kb * 32);
                WMMA_BF16(acc, afr1[kb], bm);
            }
#pragma unroll
            for (int r = 0; r < 8; r++) {
                int row = wave * 16 + r + 8 * half;
                int col = c * 128 + nt * 16 + nn;
                float v = acc[r] + fc1_b[col];
                v = 0.5f * v * (1.0f + erff(v * 0.70710678118f));
                Hs[row * 128 + nt * 16 + nn] = f2bf(v);
            }
        }
        __syncthreads();
        // gemm2 partial: out += H_chunk @ fc2T[chunk,:]
        v16bf afr2[4];
#pragma unroll
        for (int kk = 0; kk < 4; kk++) afr2[kk] = fragA(Hs, 128, wave * 16, kk * 32);
        for (int nt = 0; nt < 16; nt++) {
            v8f acc = acc2[nt];
#pragma unroll
            for (int kk = 0; kk < 4; kk++) {
                v16bf bm = fragBt(fc2T, 1024, nt * 16, c * 128 + kk * 32);
                WMMA_BF16(acc, afr2[kk], bm);
            }
            acc2[nt] = acc;
        }
        __syncthreads();
    }

    // epilogue: += fc2_b + residual (in place)
    for (int nt = 0; nt < 16; nt++) {
#pragma unroll
        for (int r = 0; r < 8; r++) {
            int row = wave * 16 + r + 8 * half;
            int col = nt * 16 + nn;
            size_t gi = (rowBase + row) * C_ + col;
            out[gi] += acc2[nt][r] + fc2_b[col];
        }
    }
}

// ---- launcher ----------------------------------------------------------
extern "C" void kernel_launch(void* const* d_in, const int* in_sizes, int n_in,
                              void* d_out, int out_size, void* d_ws, size_t ws_size,
                              hipStream_t stream) {
    const float* x      = (const float*)d_in[0];
    const float* n1w    = (const float*)d_in[1];
    const float* n1b    = (const float*)d_in[2];
    const float* qkv_w  = (const float*)d_in[3];
    const float* qkv_b  = (const float*)d_in[4];
    const float* bias_t = (const float*)d_in[5];
    const float* proj_w = (const float*)d_in[6];
    const float* proj_b = (const float*)d_in[7];
    const float* n2w    = (const float*)d_in[8];
    const float* n2b    = (const float*)d_in[9];
    const float* fc1_w  = (const float*)d_in[10];
    const float* fc1_b  = (const float*)d_in[11];
    const float* fc2_w  = (const float*)d_in[12];
    const float* fc2_b  = (const float*)d_in[13];
    float* out = (float*)d_out;

    unsigned char* ws = (unsigned char*)d_ws;
    unsigned short* qkvT  = (unsigned short*)(ws + 0);           // 768*256*2  = 393216
    unsigned short* projT = (unsigned short*)(ws + 393216);      // 256*256*2  = 131072
    unsigned short* fc1T  = (unsigned short*)(ws + 524288);      // 1024*256*2 = 524288
    unsigned short* fc2T  = (unsigned short*)(ws + 1048576);     // 256*1024*2 = 524288
    unsigned short* xw    = (unsigned short*)(ws + 1572864);     // 2048*64*256*2 = 64MB
    unsigned short* ow    = (unsigned short*)(ws + 1572864 + 67108864);

    hipLaunchKernelGGL(k_convert_weights, dim3(512), dim3(256), 0, stream,
                       qkv_w, proj_w, fc1_w, fc2_w, qkvT, projT, fc1T, fc2T);
    hipLaunchKernelGGL(k_ln1_partition, dim3(WINS_), dim3(256), 0, stream,
                       x, n1w, n1b, xw);
    hipLaunchKernelGGL(k_window_attn, dim3(WINS_, NH_), dim3(128), 0, stream,
                       xw, qkvT, qkv_b, bias_t, ow);
    hipLaunchKernelGGL(k_proj, dim3(WINS_), dim3(128), 0, stream,
                       ow, projT, proj_b, x, out);
    hipLaunchKernelGGL(k_mlp, dim3(NROWS_ / 64), dim3(128), 0, stream,
                       out, n2w, n2b, fc1T, fc1_b, fc2T, fc2_b);
}
